// Weighting_52364241273284
// MI455X (gfx1250) — matrix operand order repaired and checked by reading, not compiled, not run on hardware
//
#include <hip/hip_runtime.h>
#include <hip/hip_bf16.h>

// ---------------------------------------------------------------------------
// Shapes (fixed by the reference)
// ---------------------------------------------------------------------------
#define BB   8
#define CC   512
#define NN   1024          // H*W
#define BN   8192          // B*N tokens
#define CQK  64
#define LL   10
#define EPSV 1e-5f

typedef __attribute__((ext_vector_type(16))) __bf16 v16bf;
typedef __attribute__((ext_vector_type(8)))  float  v8f;

// ---------------------------------------------------------------------------
// Tensor Data Mover availability / arity selection
// ---------------------------------------------------------------------------
#if defined(__has_builtin)
#  if __has_builtin(__builtin_amdgcn_tensor_load_to_lds)
#    define CDNA5_HAS_TDM 1
#  endif
#endif
#ifndef CDNA5_HAS_TDM
#  define CDNA5_HAS_TDM 0
#endif
#if CDNA5_HAS_TDM
#  if defined(__has_include)
#    if __has_include(<hip/amd_detail/amd_gfx1250_TDM.h>)
#      define CDNA5_TDM_6ARG 1
#    else
#      define CDNA5_TDM_6ARG 0
#    endif
#  else
#    define CDNA5_TDM_6ARG 0
#  endif
#endif

typedef __attribute__((ext_vector_type(4))) unsigned int tdm_u32x4;
typedef __attribute__((ext_vector_type(8))) int          tdm_i32x8;
typedef __attribute__((ext_vector_type(4))) int          tdm_i32x4;

__device__ __forceinline__ unsigned short f2bf(float f) {
    unsigned int u = __float_as_uint(f);
    unsigned int r = u + 0x7FFFu + ((u >> 16) & 1u);   // round-to-nearest-even
    return (unsigned short)(r >> 16);
}

// ---------------------------------------------------------------------------
// f32 -> bf16 raw convert (weights)
// ---------------------------------------------------------------------------
__global__ __launch_bounds__(256) void cvt_bf16_kernel(const float* __restrict__ in,
                                                       unsigned short* __restrict__ out,
                                                       int n) {
    int i = blockIdx.x * 256 + threadIdx.x;
    if (i < n) out[i] = f2bf(in[i]);
}

// ---------------------------------------------------------------------------
// [B,C,N] f32 -> [B*N, C] bf16 (tiled transpose)
// ---------------------------------------------------------------------------
__global__ __launch_bounds__(256) void tokens_kernel(const float* __restrict__ x,
                                                     unsigned short* __restrict__ Xt) {
    __shared__ float tile[32][33];
    int b  = blockIdx.z;
    int c0 = blockIdx.y * 32;
    int n0 = blockIdx.x * 32;
#pragma unroll
    for (int i = 0; i < 32; i += 8) {
        int c = c0 + threadIdx.y + i;
        int n = n0 + threadIdx.x;
        tile[threadIdx.y + i][threadIdx.x] = x[((long long)b * CC + c) * NN + n];
    }
    __syncthreads();
#pragma unroll
    for (int i = 0; i < 32; i += 8) {
        int n = n0 + threadIdx.y + i;
        int c = c0 + threadIdx.x;
        Xt[((long long)b * NN + n) * CC + c] = f2bf(tile[threadIdx.x][threadIdx.y + i]);
    }
}

// ---------------------------------------------------------------------------
// Generic bf16 WMMA GEMM:  C[M,N] = A[M,K] * B[N,K]^T (+bias), batched
// Block tile 128x128, K-step 64, 8 waves (4x2), each wave 32x64 of C.
// Tile staging via Tensor Data Mover (double-buffered LDS) when available.
// biasMode: 0 none, 1 per-col (bias[n]), 2 per-row (bias[m]).
// ---------------------------------------------------------------------------
#define TM 128
#define TN 128
#define TK 64

union FragU {
    v16bf v;
    unsigned int u[8];
    uint4 q[2];
};

#if CDNA5_HAS_TDM
// Issue a 2D tile DMA: rows x TK bf16 elements, row stride = stride_elems.
// OOB rows/cols (beyond tensor_dim) are zero-filled by the TDM.
__device__ __forceinline__ void tdm_tile_load(const unsigned short* gsrc,
                                              unsigned lds_byte_off,
                                              unsigned rows_avail,
                                              unsigned k_avail,
                                              unsigned long long stride_elems) {
    unsigned long long ga = (unsigned long long)(size_t)gsrc;
    tdm_u32x4 g0;
    g0[0] = 1u;                                            // count=1, user D#
    g0[1] = lds_byte_off;                                  // lds_addr
    g0[2] = (unsigned)ga;                                  // global_addr[31:0]
    g0[3] = (unsigned)((ga >> 32) & 0x01FFFFFFu) | (2u << 30);  // addr[56:32] | type=2
    tdm_i32x8 g1;
    g1[0] = (int)(1u << 16);                               // data_size=1 -> 2 bytes
    g1[1] = (int)((k_avail & 0xFFFFu) << 16);              // tensor_dim0[15:0]
    g1[2] = (int)((k_avail >> 16) | ((rows_avail & 0xFFFFu) << 16));   // dim0 hi | dim1 lo
    g1[3] = (int)((rows_avail >> 16) | ((unsigned)TK << 16));          // dim1 hi | tile_dim0
    g1[4] = (int)TM;                                       // tile_dim1 = 128 rows
    g1[5] = (int)(stride_elems & 0xFFFFFFFFull);           // tensor_dim0_stride lo
    g1[6] = (int)((stride_elems >> 32) & 0xFFFFull);       // stride hi
    g1[7] = 0;
    tdm_i32x4 z4; z4[0] = z4[1] = z4[2] = z4[3] = 0;
#if CDNA5_TDM_6ARG
    tdm_i32x8 z8;
#pragma unroll
    for (int q = 0; q < 8; ++q) z8[q] = 0;
    __builtin_amdgcn_tensor_load_to_lds(g0, g1, z4, z4, z8, 0);
#else
    __builtin_amdgcn_tensor_load_to_lds(g0, g1, z4, z4, 0);
#endif
}
#endif

__global__ __launch_bounds__(256) void gemm_bf16_wmma(
    const unsigned short* __restrict__ A, int lda, long long sA,
    const unsigned short* __restrict__ B, int ldb, long long sB,
    void* __restrict__ Cout, int ldc, long long sC,
    const float* __restrict__ bias, int biasMode,
    int M, int N, int K, int outBf16)
{
    const int bz   = blockIdx.z;
    const unsigned short* Ab = A + (long long)bz * sA;
    const unsigned short* Bb = B + (long long)bz * sB;

    const int tid  = threadIdx.x;
    const int wave = tid >> 5;
    const int lane = tid & 31;
    const int wm   = wave & 3;        // 0..3 -> 32-row slab
    const int wn   = wave >> 2;       // 0..1 -> 64-col slab
    const int half = lane >> 4;
    const int lmod = lane & 15;

    const int m0 = blockIdx.x * TM;
    const int n0 = blockIdx.y * TN;

    v8f acc[2][4];
#pragma unroll
    for (int i = 0; i < 2; ++i)
#pragma unroll
        for (int j = 0; j < 4; ++j) {
            v8f z = {0.f, 0.f, 0.f, 0.f, 0.f, 0.f, 0.f, 0.f};
            acc[i][j] = z;
        }

#if CDNA5_HAS_TDM
    // ---------------- TDM path: DMA staging, double-buffered ----------------
    __shared__ __align__(16) unsigned short lA[2][TM * TK];
    __shared__ __align__(16) unsigned short lB[2][TN * TK];

    // prologue: stage k-tile 0
    if (wave == 0)
        tdm_tile_load(Bb + (size_t)n0 * ldb, (unsigned)(size_t)&lB[0][0],
                      (unsigned)(N - n0), (unsigned)K, (unsigned long long)ldb);
    else if (wave == 1)
        tdm_tile_load(Ab + (size_t)m0 * lda, (unsigned)(size_t)&lA[0][0],
                      (unsigned)(M - m0), (unsigned)K, (unsigned long long)lda);
    if (wave < 2) __builtin_amdgcn_s_wait_tensorcnt(0);
    __syncthreads();

    int cur = 0;
    for (int kt = 0; kt < K; kt += TK) {
        // kick off DMA for the next k-tile into the other buffer
        if (kt + TK < K) {
            if (wave == 0)
                tdm_tile_load(Bb + (size_t)n0 * ldb + kt + TK,
                              (unsigned)(size_t)&lB[cur ^ 1][0],
                              (unsigned)(N - n0), (unsigned)(K - kt - TK),
                              (unsigned long long)ldb);
            else if (wave == 1)
                tdm_tile_load(Ab + (size_t)m0 * lda + kt + TK,
                              (unsigned)(size_t)&lA[cur ^ 1][0],
                              (unsigned)(M - m0), (unsigned)(K - kt - TK),
                              (unsigned long long)lda);
        }
        const unsigned short* curA = &lA[cur][0];
        const unsigned short* curB = &lB[cur][0];
#else
    // ---------------- fallback: cooperative vector-load staging -------------
    __shared__ __align__(16) unsigned short lA[TM * TK];
    __shared__ __align__(16) unsigned short lB[TN * TK];
    const int ar = tid >> 1;              // 0..127
    const int ac = (tid & 1) * 32;        // 0 or 32 (32 elems = 4 uint4 each)

    for (int kt = 0; kt < K; kt += TK) {
        {
            const uint4* gp =
                reinterpret_cast<const uint4*>(Ab + (size_t)(m0 + ar) * lda + kt + ac);
            uint4 a0 = gp[0], a1 = gp[1], a2 = gp[2], a3 = gp[3];
            uint4* sp = reinterpret_cast<uint4*>(&lA[ar * TK + ac]);
            sp[0] = a0; sp[1] = a1; sp[2] = a2; sp[3] = a3;
            if (kt + TK < K)
                __builtin_prefetch(Ab + (size_t)(m0 + ar) * lda + kt + TK + ac, 0, 1);
        }
        {
            uint4 b0, b1, b2, b3;
            if (n0 + ar < N) {
                const uint4* gp =
                    reinterpret_cast<const uint4*>(Bb + (size_t)(n0 + ar) * ldb + kt + ac);
                b0 = gp[0]; b1 = gp[1]; b2 = gp[2]; b3 = gp[3];
            } else {
                b0.x = b0.y = b0.z = b0.w = 0u;
                b1 = b0; b2 = b0; b3 = b0;
            }
            uint4* sp = reinterpret_cast<uint4*>(&lB[ar * TK + ac]);
            sp[0] = b0; sp[1] = b1; sp[2] = b2; sp[3] = b3;
        }
        __syncthreads();
        const unsigned short* curA = &lA[0];
        const unsigned short* curB = &lB[0];
#endif

        // ---- fragments + WMMA: 2 k-substeps of 32, 8 WMMAs each ----
#pragma unroll
        for (int kk = 0; kk < TK; kk += 32) {
            FragU afr[2];
#pragma unroll
            for (int i = 0; i < 2; ++i) {
                const int am = wm * 32 + i * 16 + lmod;   // tile row (M)
#pragma unroll
                for (int v = 0; v < 8; ++v) {
                    const int k0 = kk + half * 16 + v * 2;
                    afr[i].u[v] =
                        *reinterpret_cast<const unsigned int*>(&curA[am * TK + k0]);
                }
            }
            FragU bfr[4];
#pragma unroll
            for (int j = 0; j < 4; ++j) {
                const int bn = wn * 64 + j * 16 + lmod;   // tile col (N)
                const uint4* p =
                    reinterpret_cast<const uint4*>(&curB[bn * TK + kk + half * 16]);
                bfr[j].q[0] = p[0];
                bfr[j].q[1] = p[1];
            }
#pragma unroll
            for (int i = 0; i < 2; ++i)
#pragma unroll
                for (int j = 0; j < 4; ++j)
                    acc[i][j] = __builtin_amdgcn_wmma_f32_16x16x32_bf16(
                        false, afr[i].v, false, bfr[j].v,
                        (short)0, acc[i][j], false, false);
        }

#if CDNA5_HAS_TDM
        if (wave < 2) __builtin_amdgcn_s_wait_tensorcnt(0);
        __syncthreads();
        cur ^= 1;
    }
#else
        __syncthreads();
    }
#endif

    // --- epilogue ---
#pragma unroll
    for (int i = 0; i < 2; ++i) {
#pragma unroll
        for (int j = 0; j < 4; ++j) {
            const int gm0 = m0 + wm * 32 + i * 16 + half * 8;
            const int gn  = n0 + wn * 64 + j * 16 + lmod;
            if (gn < N) {
#pragma unroll
                for (int r = 0; r < 8; ++r) {
                    const int gm = gm0 + r;
                    float v = acc[i][j][r];
                    if (biasMode == 1)      v += bias[gn];
                    else if (biasMode == 2) v += bias[gm];
                    const size_t idx = (size_t)bz * sC + (size_t)gm * ldc + gn;
                    if (outBf16)
                        ((unsigned short*)Cout)[idx] = f2bf(v);
                    else
                        ((float*)Cout)[idx] = v;
                }
            }
        }
    }
}

// ---------------------------------------------------------------------------
// Row softmax over S[B,N,N] fp32 -> P bf16 and P^T bf16
// ---------------------------------------------------------------------------
__global__ __launch_bounds__(256) void softmax_kernel(const float* __restrict__ S,
                                                      unsigned short* __restrict__ P,
                                                      unsigned short* __restrict__ PT) {
    const long long row = blockIdx.x;            // b*N + n
    const long long b   = row >> 10;
    const long long n   = row & (NN - 1);
    const float* s = S + row * NN;
    __shared__ float red[256];
    const int tid = threadIdx.x;

    float lmax = -3.0e38f;
    for (int m = tid; m < NN; m += 256) lmax = fmaxf(lmax, s[m]);
    red[tid] = lmax;
    __syncthreads();
    for (int st = 128; st > 0; st >>= 1) {
        if (tid < st) red[tid] = fmaxf(red[tid], red[tid + st]);
        __syncthreads();
    }
    const float mx = red[0];
    __syncthreads();

    float lsum = 0.f;
    for (int m = tid; m < NN; m += 256) lsum += __expf(s[m] - mx);
    red[tid] = lsum;
    __syncthreads();
    for (int st = 128; st > 0; st >>= 1) {
        if (tid < st) red[tid] += red[tid + st];
        __syncthreads();
    }
    const float inv = 1.f / red[0];

    for (int m = tid; m < NN; m += 256) {
        const unsigned short pb = f2bf(__expf(s[m] - mx) * inv);
        P[row * NN + m]             = pb;
        PT[(b * NN + m) * NN + n]   = pb;   // scattered, L2-resident
    }
}

// ---------------------------------------------------------------------------
// hidden[L,B,C,H,W] -> out[B,C] = sum_l softmax(hw)[l] * mean_n hidden[l,b,c,:]
// ---------------------------------------------------------------------------
__global__ __launch_bounds__(256) void hidden_mean_kernel(const float* __restrict__ hid,
                                                          const float* __restrict__ hw,
                                                          float* __restrict__ outBC) {
    const int c = blockIdx.x;
    const int b = blockIdx.y;
    const int tid = threadIdx.x;

    float e[LL];
    float mx = hw[0];
#pragma unroll
    for (int l = 1; l < LL; ++l) mx = fmaxf(mx, hw[l]);
    float s = 0.f;
#pragma unroll
    for (int l = 0; l < LL; ++l) { e[l] = __expf(hw[l] - mx); s += e[l]; }
    const float invS = 1.f / (s * (float)NN);

    float accv = 0.f;
#pragma unroll
    for (int l = 0; l < LL; ++l) {
        const float* p = hid + (((long long)l * BB + b) * CC + c) * NN;
        float ls = 0.f;
        for (int n = tid; n < NN; n += 256) ls += p[n];
        accv += e[l] * ls;
    }
    __shared__ float red[256];
    red[tid] = accv;
    __syncthreads();
    for (int st = 128; st > 0; st >>= 1) {
        if (tid < st) red[tid] += red[tid + st];
        __syncthreads();
    }
    if (tid == 0) outBC[(long long)b * CC + c] = red[0] * invS;
}

// ---------------------------------------------------------------------------
// gamma MLP: [B,C] -> relu(v W1^T + b1)[B,64] -> (@ W2^T + b2)[B] ; one block
// ---------------------------------------------------------------------------
__global__ __launch_bounds__(256) void gamma_kernel(const float* __restrict__ hwv,
                                                    const float* __restrict__ W1,
                                                    const float* __restrict__ b1,
                                                    const float* __restrict__ W2,
                                                    const float* __restrict__ b2,
                                                    float* __restrict__ gamma) {
    __shared__ float h[BB * 64];
    for (int idx = threadIdx.x; idx < BB * 64; idx += 256) {
        const int b = idx >> 6, j = idx & 63;
        float a = b1[j];
        for (int c = 0; c < CC; ++c) a += hwv[b * CC + c] * W1[j * CC + c];
        h[idx] = fmaxf(a, 0.f);
    }
    __syncthreads();
    if (threadIdx.x < BB) {
        float a = b2[0];
#pragma unroll
        for (int j = 0; j < 64; ++j) a += h[threadIdx.x * 64 + j] * W2[j];
        gamma[threadIdx.x] = a;
    }
}

// ---------------------------------------------------------------------------
// G[t,c] = bf16(gamma[batch(t)] * O[t,c]) ; token-major [BN, C]
// ---------------------------------------------------------------------------
__global__ __launch_bounds__(256) void scale_bf16_kernel(const float* __restrict__ O,
                                                         const float* __restrict__ gamma,
                                                         unsigned short* __restrict__ G) {
    const long long i = (long long)blockIdx.x * 256 + threadIdx.x;   // < BN*C
    const int b = (int)(i / ((long long)NN * CC));
    G[i] = f2bf(O[i] * gamma[b]);
}

// ---------------------------------------------------------------------------
// Per-channel sum / sumsq over token-major Z[BN, C] ; grid = C blocks
// ---------------------------------------------------------------------------
__global__ __launch_bounds__(256) void bnstats_kernel(const float* __restrict__ Z,
                                                      float* __restrict__ stats) {
    const int c = blockIdx.x;
    const int tid = threadIdx.x;
    float s = 0.f, sq = 0.f;
    for (int t = tid; t < BN; t += 256) {
        const float v = Z[(long long)t * CC + c];
        s += v;
        sq += v * v;
    }
    __shared__ float r0[256], r1[256];
    r0[tid] = s;
    r1[tid] = sq;
    __syncthreads();
    for (int st = 128; st > 0; st >>= 1) {
        if (tid < st) { r0[tid] += r0[tid + st]; r1[tid] += r1[tid + st]; }
        __syncthreads();
    }
    if (tid == 0) { stats[c] = r0[0]; stats[CC + c] = r1[0]; }
}

// ---------------------------------------------------------------------------
// BN(train) + relu -> bf16 token-major
// ---------------------------------------------------------------------------
__global__ __launch_bounds__(256) void bnrelu_bf16_kernel(const float* __restrict__ Z,
                                                          const float* __restrict__ stats,
                                                          const float* __restrict__ g,
                                                          const float* __restrict__ bb,
                                                          unsigned short* __restrict__ out) {
    const long long i = (long long)blockIdx.x * 256 + threadIdx.x;
    const int c = (int)(i & (CC - 1));
    const float cnt  = (float)BN;
    const float mean = stats[c] / cnt;
    const float var  = stats[CC + c] / cnt - mean * mean;
    const float inv  = rsqrtf(var + EPSV);
    const float v = (Z[i] - mean) * (g[c] * inv) + bb[c];
    out[i] = f2bf(fmaxf(v, 0.f));
}

// ---------------------------------------------------------------------------
// Final: out[b,c,n] = relu( BN(Z2)[t,c] + skip[b,c,n] ), t = b*N + n
// ---------------------------------------------------------------------------
__global__ __launch_bounds__(256) void final_kernel(const float* __restrict__ Z2,
                                                    const float* __restrict__ stats,
                                                    const float* __restrict__ g,
                                                    const float* __restrict__ bb,
                                                    const float* __restrict__ skip,
                                                    float* __restrict__ out) {
    const long long i = (long long)blockIdx.x * 256 + threadIdx.x;   // < BN*C
    const int c = (int)(i & (CC - 1));
    const long long t = i >> 9;           // C == 512
    const int b = (int)(t >> 10);
    const int n = (int)(t & (NN - 1));
    const float cnt  = (float)BN;
    const float mean = stats[c] / cnt;
    const float var  = stats[CC + c] / cnt - mean * mean;
    const float inv  = rsqrtf(var + EPSV);
    const float r = (Z2[i] - mean) * (g[c] * inv) + bb[c];
    const long long xi = ((long long)b * CC + c) * NN + n;
    out[xi] = fmaxf(r + skip[xi], 0.f);
}

// ---------------------------------------------------------------------------
// host launch
// ---------------------------------------------------------------------------
extern "C" void kernel_launch(void* const* d_in, const int* in_sizes, int n_in,
                              void* d_out, int out_size, void* d_ws, size_t ws_size,
                              hipStream_t stream) {
    const float* x        = (const float*)d_in[0];
    const float* y        = (const float*)d_in[1];
    const float* x_hidden = (const float*)d_in[2];
    const float* y_hidden = (const float*)d_in[3];
    const float* qW  = (const float*)d_in[4];
    const float* qb  = (const float*)d_in[5];
    const float* kW  = (const float*)d_in[6];
    const float* kb  = (const float*)d_in[7];
    const float* v1W = (const float*)d_in[8];
    const float* v1b = (const float*)d_in[9];
    const float* v2W = (const float*)d_in[10];
    const float* v2b = (const float*)d_in[11];
    const float* hw1 = (const float*)d_in[12];
    const float* hw2 = (const float*)d_in[13];
    const float* g1W1 = (const float*)d_in[14];
    const float* g1b1 = (const float*)d_in[15];
    const float* g1W2 = (const float*)d_in[16];
    const float* g1b2 = (const float*)d_in[17];
    const float* g2W1 = (const float*)d_in[18];
    const float* g2b1 = (const float*)d_in[19];
    const float* g2W2 = (const float*)d_in[20];
    const float* g2b2 = (const float*)d_in[21];
    const float* rW1 = (const float*)d_in[22];
    const float* rb1 = (const float*)d_in[23];
    const float* bn1g = (const float*)d_in[24];
    const float* bn1b = (const float*)d_in[25];
    const float* rW2 = (const float*)d_in[26];
    const float* rb2 = (const float*)d_in[27];
    const float* bn2g = (const float*)d_in[28];
    const float* bn2b = (const float*)d_in[29];

    float* out = (float*)d_out;

    // ---- workspace arena ----
    char* W = (char*)d_ws;
    size_t off = 0;
    auto take = [&](size_t bytes) -> char* {
        char* p = W + off;
        off += (bytes + 255) & ~(size_t)255;
        return p;
    };
    unsigned short* Xt   = (unsigned short*)take((size_t)BN * CC * 2);
    unsigned short* Yt   = (unsigned short*)take((size_t)BN * CC * 2);
    unsigned short* qWb  = (unsigned short*)take((size_t)CQK * CC * 2);
    unsigned short* kWb  = (unsigned short*)take((size_t)CQK * CC * 2);
    unsigned short* v1Wb = (unsigned short*)take((size_t)CC * CC * 2);
    unsigned short* v2Wb = (unsigned short*)take((size_t)CC * CC * 2);
    unsigned short* rW1b = (unsigned short*)take((size_t)CC * CC * 2);
    unsigned short* rW2b = (unsigned short*)take((size_t)CC * CC * 2);
    unsigned short* Qb   = (unsigned short*)take((size_t)BN * CQK * 2);
    unsigned short* Kb   = (unsigned short*)take((size_t)BN * CQK * 2);
    unsigned short* V1c  = (unsigned short*)take((size_t)CC * BN * 2);
    unsigned short* V2c  = (unsigned short*)take((size_t)CC * BN * 2);
    float* xhw  = (float*)take((size_t)BB * CC * 4);
    float* yhw  = (float*)take((size_t)BB * CC * 4);
    float* gam1 = (float*)take(BB * 4);
    float* gam2 = (float*)take(BB * 4);
    float* S    = (float*)take((size_t)BB * NN * NN * 4);
    unsigned short* P  = (unsigned short*)take((size_t)BB * NN * NN * 2);
    unsigned short* PT = (unsigned short*)take((size_t)BB * NN * NN * 2);
    float* O1 = (float*)take((size_t)BN * CC * 4);
    float* O2 = (float*)take((size_t)BN * CC * 4);
    unsigned short* G1 = (unsigned short*)take((size_t)BN * CC * 2);
    unsigned short* G2 = (unsigned short*)take((size_t)BN * CC * 2);
    float* Z1 = (float*)take((size_t)BN * CC * 4);
    unsigned short* Z1n = (unsigned short*)take((size_t)BN * CC * 2);
    float* Z2 = (float*)take((size_t)BN * CC * 4);
    float* st1 = (float*)take(2 * CC * 4);
    float* st2 = (float*)take(2 * CC * 4);

    // ---- 1. weights -> bf16 ----
    cvt_bf16_kernel<<<(CQK * CC + 255) / 256, 256, 0, stream>>>(qW, qWb, CQK * CC);
    cvt_bf16_kernel<<<(CQK * CC + 255) / 256, 256, 0, stream>>>(kW, kWb, CQK * CC);
    cvt_bf16_kernel<<<(CC * CC + 255) / 256, 256, 0, stream>>>(v1W, v1Wb, CC * CC);
    cvt_bf16_kernel<<<(CC * CC + 255) / 256, 256, 0, stream>>>(v2W, v2Wb, CC * CC);
    cvt_bf16_kernel<<<(CC * CC + 255) / 256, 256, 0, stream>>>(rW1, rW1b, CC * CC);
    cvt_bf16_kernel<<<(CC * CC + 255) / 256, 256, 0, stream>>>(rW2, rW2b, CC * CC);

    // ---- 2. x,y -> token-major bf16 ----
    {
        dim3 tb(32, 8), tg(NN / 32, CC / 32, BB);
        tokens_kernel<<<tg, tb, 0, stream>>>(x, Xt);
        tokens_kernel<<<tg, tb, 0, stream>>>(y, Yt);
    }

    // ---- 3. hidden means + gamma MLPs (bandwidth-bound path) ----
    hidden_mean_kernel<<<dim3(CC, BB), 256, 0, stream>>>(x_hidden, hw1, xhw);
    hidden_mean_kernel<<<dim3(CC, BB), 256, 0, stream>>>(y_hidden, hw2, yhw);
    gamma_kernel<<<1, 256, 0, stream>>>(xhw, g1W1, g1b1, g1W2, g1b2, gam1);
    gamma_kernel<<<1, 256, 0, stream>>>(yhw, g2W1, g2b1, g2W2, g2b2, gam2);

    // ---- 4. projections ----
    gemm_bf16_wmma<<<dim3(BN / TM, 1, 1), 256, 0, stream>>>(
        Xt, CC, 0, qWb, CC, 0, Qb, CQK, 0, qb, 1, BN, CQK, CC, 1);
    gemm_bf16_wmma<<<dim3(BN / TM, 1, 1), 256, 0, stream>>>(
        Yt, CC, 0, kWb, CC, 0, Kb, CQK, 0, kb, 1, BN, CQK, CC, 1);
    gemm_bf16_wmma<<<dim3(CC / TM, BN / TN, 1), 256, 0, stream>>>(
        v1Wb, CC, 0, Xt, CC, 0, V1c, BN, 0, v1b, 2, CC, BN, CC, 1);
    gemm_bf16_wmma<<<dim3(CC / TM, BN / TN, 1), 256, 0, stream>>>(
        v2Wb, CC, 0, Yt, CC, 0, V2c, BN, 0, v2b, 2, CC, BN, CC, 1);

    // ---- 5. attention logits S_b = Q_b @ K_b^T (batched, fp32 out) ----
    gemm_bf16_wmma<<<dim3(NN / TM, NN / TN, BB), 256, 0, stream>>>(
        Qb, CQK, (long long)NN * CQK, Kb, CQK, (long long)NN * CQK,
        S, NN, (long long)NN * NN, nullptr, 0, NN, NN, CQK, 0);

    // ---- 6. softmax -> P, P^T (bf16) ----
    softmax_kernel<<<BB * NN, 256, 0, stream>>>(S, P, PT);

    // ---- 7. attention apply ----
    gemm_bf16_wmma<<<dim3(NN / TM, CC / TN, BB), 256, 0, stream>>>(
        P, NN, (long long)NN * NN, V1c, BN, (long long)NN,
        O1, CC, (long long)NN * CC, nullptr, 0, NN, CC, NN, 0);
    gemm_bf16_wmma<<<dim3(NN / TM, CC / TN, BB), 256, 0, stream>>>(
        PT, NN, (long long)NN * NN, V2c, BN, (long long)NN,
        O2, CC, (long long)NN * CC, nullptr, 0, NN, CC, NN, 0);

    // ---- 8. gamma scale -> bf16 ----
    scale_bf16_kernel<<<(BN * CC) / 256, 256, 0, stream>>>(O1, gam1, G1);
    scale_bf16_kernel<<<(BN * CC) / 256, 256, 0, stream>>>(O2, gam2, G2);

    // ---- 9. residual branch, path 1 then path 2 (shared scratch) ----
    for (int path = 0; path < 2; ++path) {
        const unsigned short* Gin = path == 0 ? G1 : G2;
        const float* skip = path == 0 ? x : y;
        float* outp = out + (size_t)path * BB * CC * NN;

        gemm_bf16_wmma<<<dim3(BN / TM, CC / TN, 1), 256, 0, stream>>>(
            Gin, CC, 0, rW1b, CC, 0, Z1, CC, 0, rb1, 1, BN, CC, CC, 0);
        bnstats_kernel<<<CC, 256, 0, stream>>>(Z1, st1);
        bnrelu_bf16_kernel<<<(BN * CC) / 256, 256, 0, stream>>>(Z1, st1, bn1g, bn1b, Z1n);
        gemm_bf16_wmma<<<dim3(BN / TM, CC / TN, 1), 256, 0, stream>>>(
            Z1n, CC, 0, rW2b, CC, 0, Z2, CC, 0, rb2, 1, BN, CC, CC, 0);
        bnstats_kernel<<<CC, 256, 0, stream>>>(Z2, st2);
        final_kernel<<<(BN * CC) / 256, 256, 0, stream>>>(Z2, st2, bn2g, bn2b, skip, outp);
    }
    (void)in_sizes; (void)n_in; (void)out_size; (void)ws_size;
}
